// VectorQuantizerEMA_62577673503199
// MI455X (gfx1250) — matrix operand order repaired and checked by reading, not compiled, not run on hardware
//
#include <hip/hip_runtime.h>

typedef float  v2f  __attribute__((ext_vector_type(2)));
typedef float  v4f  __attribute__((ext_vector_type(4)));
typedef float  v8f  __attribute__((ext_vector_type(8)));
typedef __bf16 v8bf  __attribute__((ext_vector_type(8)));
typedef __bf16 v16bf __attribute__((ext_vector_type(16)));
typedef unsigned int v4u __attribute__((ext_vector_type(4)));
typedef int    v4i  __attribute__((vector_size(16)));   // matches async-LDS builtin pointee

#define VQ_N   64
#define VQ_D   256
#define VQ_T   2048
#define VQ_K   1024
#define VQ_R   (VQ_N * VQ_T)                 /* 131072 flattened rows */
#define VQ_E   ((size_t)VQ_N * VQ_D * VQ_T)  /* 33554432 elements     */
#define MTILE  64                            /* rows per workgroup    */
#define NCHUNK 64                            /* codewords per chunk   */
#define DSTR   264                           /* LDS row stride (bf16) */

#define AS1 __attribute__((address_space(1)))
#define AS3 __attribute__((address_space(3)))

#if __has_builtin(__builtin_amdgcn_global_load_async_to_lds_b128)
#define VQ_ASYNC 1
#else
#define VQ_ASYNC 0
#endif

__device__ __forceinline__ unsigned short f2bf(float f) {   // RNE f32 -> bf16
    unsigned u = __float_as_uint(f);
    return (unsigned short)((u + 0x7fffu + ((u >> 16) & 1u)) >> 16);
}

__device__ __forceinline__ void vq_wait_async0() {
#if __has_builtin(__builtin_amdgcn_s_wait_asynccnt)
    __builtin_amdgcn_s_wait_asynccnt(0);
#else
    asm volatile("s_wait_asynccnt 0x0" ::: "memory");
#endif
}

// ---------------------------------------------------------------------------
// Kernel 1: codebook norms, zero histogram, and bf16 hi/lo planes of codebook.
//   cb = cbh + cbl with cbh = bf16(cb), cbl = bf16(cb - cbh)
// ---------------------------------------------------------------------------
__global__ __launch_bounds__(256) void vq_prep_kernel(const float* __restrict__ cb,
                                                      float* __restrict__ enorm,
                                                      int* __restrict__ counts,
                                                      unsigned short* __restrict__ cbh,
                                                      unsigned short* __restrict__ cbl) {
    int k = blockIdx.x * blockDim.x + threadIdx.x;
    if (k < VQ_K) {
        float s = 0.f;
        #pragma unroll 4
        for (int d = 0; d < VQ_D; ++d) {
            float v = cb[(size_t)k * VQ_D + d];
            unsigned short h = f2bf(v);
            float hf = __uint_as_float((unsigned)h << 16);
            cbh[(size_t)k * VQ_D + d] = h;
            cbl[(size_t)k * VQ_D + d] = f2bf(v - hf);
            s += v * v;
        }
        enorm[k]  = s;
        counts[k] = 0;
    }
}

// ---------------------------------------------------------------------------
// Kernel 2: distance GEMM via split-bf16 V_WMMA_F32_16X16X32_BF16 + fused argmin.
//   x.e ~= xh.eh + xh.el + xl.eh   (f32 accumulate; lo*lo dropped)
//   score(r,k) = ||e_k||^2 - 2 * x_r . e_k
// Block = 128 threads (4 waves), M-tile = 64 rows, N staged in 64-wide chunks.
// Three separate accumulator sets -> 12 independent WMMA chains per K-step
// (>= 5-slot bf16 WMMA->WMMA hazard distance, no v_nop padding).
// ---------------------------------------------------------------------------
__global__ __launch_bounds__(128) void vq_argmin_kernel(const float* __restrict__ x,
                                                        const unsigned short* __restrict__ cbh,
                                                        const unsigned short* __restrict__ cbl,
                                                        const float* __restrict__ enorm,
                                                        int* __restrict__ outIdx) {
    __shared__ __align__(16) unsigned short xs_h[MTILE][DSTR];
    __shared__ __align__(16) unsigned short xs_l[MTILE][DSTR];
    __shared__ __align__(16) unsigned short cs_h[NCHUNK][DSTR];
    __shared__ __align__(16) unsigned short cs_l[NCHUNK][DSTR];

    const int tid  = threadIdx.x;
    const int wave = tid >> 5;
    const int lane = tid & 31;
    const int l16  = lane & 15;
    const int hi   = lane >> 4;     // half-wave select (ISA 16-bit fragment layout)

    const int rowBase = blockIdx.x * MTILE;
    const int n  = rowBase / VQ_T;
    const int t0 = rowBase % VQ_T;

    // Stage x tile as bf16 hi/lo planes: xs[tl][d] <- inputs[n, d, t0+tl].
    // Two adjacent d per thread -> one packed dword LDS store per plane.
    for (int i = tid; i < MTILE * (VQ_D / 2); i += 128) {
        int d2 = i >> 6;                 // d pair index
        int tl = i & 63;
        size_t base = (size_t)n * VQ_D * VQ_T + (size_t)(2 * d2) * VQ_T + (t0 + tl);
        float f0 = x[base];
        float f1 = x[base + VQ_T];
        unsigned short h0 = f2bf(f0), h1 = f2bf(f1);
        float hf0 = __uint_as_float((unsigned)h0 << 16);
        float hf1 = __uint_as_float((unsigned)h1 << 16);
        unsigned short l0 = f2bf(f0 - hf0), l1 = f2bf(f1 - hf1);
        *(unsigned int*)&xs_h[tl][2 * d2] = (unsigned)h0 | ((unsigned)h1 << 16);
        *(unsigned int*)&xs_l[tl][2 * d2] = (unsigned)l0 | ((unsigned)l1 << 16);
    }

    const int wrow = wave * 16 + l16;    // A-fragment row for this lane

    float bestv[8];
    int   besti[8];
    #pragma unroll
    for (int v = 0; v < 8; ++v) { bestv[v] = 3.0e38f; besti[v] = 0; }

    for (int nc = 0; nc < VQ_K; nc += NCHUNK) {
        __syncthreads();
        // Stage codebook chunk (bf16 planes) into LDS.
#if VQ_ASYNC
        for (int i = tid; i < NCHUNK * (VQ_D / 8); i += 128) {
            int cl = i >> 5;             // chunk row
            int c8 = i & 31;             // 8-element (16B) segment
            const unsigned short* gh = cbh + (size_t)(nc + cl) * VQ_D + c8 * 8;
            const unsigned short* gl = cbl + (size_t)(nc + cl) * VQ_D + c8 * 8;
            __builtin_amdgcn_global_load_async_to_lds_b128(
                (AS1 v4i*)gh, (AS3 v4i*)&cs_h[cl][c8 * 8], 0, 0);
            __builtin_amdgcn_global_load_async_to_lds_b128(
                (AS1 v4i*)gl, (AS3 v4i*)&cs_l[cl][c8 * 8], 0, 0);
        }
        vq_wait_async0();
#else
        for (int i = tid; i < NCHUNK * (VQ_D / 8); i += 128) {
            int cl = i >> 5;
            int c8 = i & 31;
            *(v4u*)&cs_h[cl][c8 * 8] = *(const v4u*)(cbh + (size_t)(nc + cl) * VQ_D + c8 * 8);
            *(v4u*)&cs_l[cl][c8 * 8] = *(const v4u*)(cbl + (size_t)(nc + cl) * VQ_D + c8 * 8);
        }
#endif
        __syncthreads();

        v8f accHH[4] = {};                // xh . eh
        v8f accHL[4] = {};                // xh . el
        v8f accLH[4] = {};                // xl . eh

        #pragma unroll 2
        for (int kk = 0; kk < VQ_D; kk += 32) {
            // A fragment (16x32 bf16): lanes 0-15 hold K {kk..kk+7, kk+16..kk+23},
            // lanes 16-31 hold K {kk+8..kk+15, kk+24..kk+31}.
            v8bf a0h = *(const v8bf*)&xs_h[wrow][kk + 8 * hi];
            v8bf a1h = *(const v8bf*)&xs_h[wrow][kk + 16 + 8 * hi];
            v8bf a0l = *(const v8bf*)&xs_l[wrow][kk + 8 * hi];
            v8bf a1l = *(const v8bf*)&xs_l[wrow][kk + 16 + 8 * hi];
            v16bf ah = __builtin_shufflevector(a0h, a1h, 0,1,2,3,4,5,6,7,8,9,10,11,12,13,14,15);
            v16bf al = __builtin_shufflevector(a0l, a1l, 0,1,2,3,4,5,6,7,8,9,10,11,12,13,14,15);

            #pragma unroll
            for (int j = 0; j < 4; ++j) {
                // B fragment (32x16 bf16): N = l16 (per subtile), lanes 0-15 K {kk..kk+7, kk+16..kk+23},
                // lanes 16-31 K {kk+8..kk+15, kk+24..kk+31}.
                const int col = j * 16 + l16;
                v8bf b0h = *(const v8bf*)&cs_h[col][kk + 8 * hi];
                v8bf b1h = *(const v8bf*)&cs_h[col][kk + 16 + 8 * hi];
                v8bf b0l = *(const v8bf*)&cs_l[col][kk + 8 * hi];
                v8bf b1l = *(const v8bf*)&cs_l[col][kk + 16 + 8 * hi];
                v16bf bh = __builtin_shufflevector(b0h, b1h, 0,1,2,3,4,5,6,7,8,9,10,11,12,13,14,15);
                v16bf bl = __builtin_shufflevector(b0l, b1l, 0,1,2,3,4,5,6,7,8,9,10,11,12,13,14,15);

                accHH[j] = __builtin_amdgcn_wmma_f32_16x16x32_bf16(
                    false, ah, false, bh, (short)0, accHH[j], false, false);
                accHL[j] = __builtin_amdgcn_wmma_f32_16x16x32_bf16(
                    false, ah, false, bl, (short)0, accHL[j], false, false);
                accLH[j] = __builtin_amdgcn_wmma_f32_16x16x32_bf16(
                    false, al, false, bh, (short)0, accLH[j], false, false);
            }
        }

        // Fold ||e||^2 - 2*dot into running per-lane argmin.
        // C layout: VGPR v, lanes 0-15 -> row v, lanes 16-31 -> row v+8; N = l16.
        #pragma unroll
        for (int j = 0; j < 4; ++j) {
            int   col = nc + j * 16 + l16;
            float en  = enorm[col];
            #pragma unroll
            for (int v = 0; v < 8; ++v) {
                float dot = accHH[j][v] + accHL[j][v] + accLH[j][v];
                float s   = en - 2.0f * dot;
                if (s < bestv[v]) { bestv[v] = s; besti[v] = col; }  // strict '<': first idx wins
            }
        }
    }

    // Cross-lane min-reduce within each 16-lane half (rows differ across halves).
    #pragma unroll
    for (int v = 0; v < 8; ++v) {
        float bv = bestv[v];
        int   bi = besti[v];
        #pragma unroll
        for (int m = 1; m < 16; m <<= 1) {
            float ov = __shfl_xor(bv, m, 32);
            int   oi = __shfl_xor(bi, m, 32);
            if (ov < bv || (ov == bv && oi < bi)) { bv = ov; bi = oi; }
        }
        if (l16 == 0) {
            int row = wave * 16 + v + hi * 8;
            outIdx[rowBase + row] = bi;
        }
    }
}

// ---------------------------------------------------------------------------
// Kernel 3: zero the 512MB one-hot encodings with 128-bit stores.
// ---------------------------------------------------------------------------
__global__ __launch_bounds__(256) void vq_zero_enc_kernel(v4f* __restrict__ p, long n4) {
    long i      = (long)blockIdx.x * blockDim.x + threadIdx.x;
    long stride = (long)gridDim.x * blockDim.x;
    v4f z = {0.f, 0.f, 0.f, 0.f};
    for (; i < n4; i += stride) p[i] = z;
}

// ---------------------------------------------------------------------------
// Kernel 4: scatter one-hot 1.0f + integer histogram (deterministic atomics).
// ---------------------------------------------------------------------------
__global__ __launch_bounds__(256) void vq_scatter_kernel(const int* __restrict__ idx,
                                                         float* __restrict__ enc,
                                                         int* __restrict__ counts) {
    int r = blockIdx.x * blockDim.x + threadIdx.x;   // exactly VQ_R threads
    int k = idx[r];
    enc[(size_t)r * VQ_K + k] = 1.0f;
    atomicAdd(&counts[k], 1);
}

// ---------------------------------------------------------------------------
// Kernel 5: quantized_out[n,d,t] = codebook[idx[n,t], d]  + per-block MSE partial.
// ---------------------------------------------------------------------------
__global__ __launch_bounds__(256) void vq_quantize_kernel(const float* __restrict__ x,
                                                          const float* __restrict__ cb,
                                                          const int* __restrict__ idx,
                                                          float* __restrict__ qout,
                                                          float* __restrict__ partial) {
    __shared__ float red[256];
    size_t gid = (size_t)blockIdx.x * 256 + threadIdx.x;
    int    t   = (int)(gid % VQ_T);
    size_t nd  = gid / VQ_T;
    int    d   = (int)(nd % VQ_D);
    int    n   = (int)(nd / VQ_D);

    int   k  = idx[n * VQ_T + t];
    float q  = cb[(size_t)k * VQ_D + d];
    float xv = x[gid];
    qout[gid] = q;

    float diff = q - xv;
    red[threadIdx.x] = diff * diff;
    __syncthreads();
    for (int w = 128; w > 0; w >>= 1) {
        if (threadIdx.x < w) red[threadIdx.x] += red[threadIdx.x + w];
        __syncthreads();
    }
    if (threadIdx.x == 0) partial[blockIdx.x] = red[0];
}

// ---------------------------------------------------------------------------
// Kernel 6: deterministic finalize -> loss, perplexity, reset_ratio.
// ---------------------------------------------------------------------------
__global__ __launch_bounds__(256) void vq_finalize_kernel(const float* __restrict__ partial,
                                                          int nPart,
                                                          const int* __restrict__ counts,
                                                          float* __restrict__ out) {
    __shared__ float red[256];
    int tid = threadIdx.x;

    float s = 0.f;
    for (int i = tid; i < nPart; i += 256) s += partial[i];   // fixed order
    red[tid] = s;
    __syncthreads();
    for (int w = 128; w > 0; w >>= 1) {
        if (tid < w) red[tid] += red[tid + w];
        __syncthreads();
    }
    float mseSum = red[0];
    __syncthreads();

    float h = 0.f;
    const float invR = 1.0f / (float)VQ_R;
    for (int i = tid; i < VQ_K; i += 256) {
        float p = (float)counts[i] * invR;
        h += p * __logf(p + 1e-10f);
    }
    red[tid] = h;
    __syncthreads();
    for (int w = 128; w > 0; w >>= 1) {
        if (tid < w) red[tid] += red[tid + w];
        __syncthreads();
    }

    if (tid == 0) {
        float mse = mseSum / (float)VQ_E;
        out[0]        = 2.0f * mse;       // q_latent + 1.0 * e_latent (equal in eval)
        out[1 + VQ_E] = __expf(-red[0]);  // perplexity
        out[2 + VQ_E] = 0.0f;             // reset_ratio
    }
}

// ---------------------------------------------------------------------------
// Launch
// d_out: [loss(1) | quantized_out(VQ_E) | perplexity(1) | reset_ratio(1) | enc(R*K)]
// d_ws : [idx: R i32 | counts: K i32 | enorm: K f32 | partials: 131072 f32 |
//         cbh: K*D u16 | cbl: K*D u16]   (~2.1 MB)
// ---------------------------------------------------------------------------
extern "C" void kernel_launch(void* const* d_in, const int* in_sizes, int n_in,
                              void* d_out, int out_size, void* d_ws, size_t ws_size,
                              hipStream_t stream) {
    const float* x  = (const float*)d_in[0];   // (64, 256, 2048) f32
    const float* cb = (const float*)d_in[1];   // (1024, 256) f32

    float* out  = (float*)d_out;
    float* qout = out + 1;
    float* enc  = out + 3 + VQ_E;

    int*   idx     = (int*)d_ws;
    int*   counts  = idx + VQ_R;
    float* enorm   = (float*)(counts + VQ_K);
    float* partial = enorm + VQ_K;
    unsigned short* cbh = (unsigned short*)(partial + (VQ_E / 256));
    unsigned short* cbl = cbh + (size_t)VQ_K * VQ_D;

    const int nQBlocks = (int)(VQ_E / 256);    // 131072

    vq_prep_kernel<<<(VQ_K + 255) / 256, 256, 0, stream>>>(cb, enorm, counts, cbh, cbl);
    vq_argmin_kernel<<<VQ_R / MTILE, 128, 0, stream>>>(x, cbh, cbl, enorm, idx);
    vq_zero_enc_kernel<<<8192, 256, 0, stream>>>((v4f*)enc, (long)VQ_R * VQ_K / 4);
    vq_scatter_kernel<<<VQ_R / 256, 256, 0, stream>>>(idx, enc, counts);
    vq_quantize_kernel<<<nQBlocks, 256, 0, stream>>>(x, cb, idx, qout, partial);
    vq_finalize_kernel<<<1, 256, 0, stream>>>(partial, nQBlocks, counts, out);
}
